// AttentionLSTM_85143431675959
// MI455X (gfx1250) — matrix operand order repaired and checked by reading, not compiled, not run on hardware
//
#include <hip/hip_runtime.h>
#include <hip/hip_bf16.h>

// Sizes (fixed by the reference)
#define NB   512      // batch
#define TT   32       // timesteps
#define DD   1024     // input dim
#define HH   1024     // hidden dim
#define KTOT 3072     // concatenated K = 3H (h | attn | x_t)
#define NC   4096     // 4H output columns

typedef __attribute__((ext_vector_type(16))) __bf16 v16bf;
typedef __attribute__((ext_vector_type(8)))  float  v8f;

// ---- helpers -------------------------------------------------------------

__device__ inline unsigned short f2bf(float f) {
    unsigned int u = __float_as_uint(f);
    unsigned int lsb = (u >> 16) & 1u;
    u += 0x7fffu + lsb;                 // round-to-nearest-even
    return (unsigned short)(u >> 16);
}

// A-operand fragment: lane half hb in {0,1}; K = {hb*8..hb*8+7} U {hb*8+16..hb*8+23}
__device__ inline v16bf loadA_frag(const unsigned short* row, int kk, int hb) {
    const unsigned short* p = row + kk + hb * 8;
    union { uint4 u[2]; v16bf v; } t;
    t.u[0] = *reinterpret_cast<const uint4*>(p);
    t.u[1] = *reinterpret_cast<const uint4*>(p + 16);
    return t.v;
}

// B-operand fragment: lane half hb; K = {hb*8..hb*8+7} U {hb*8+16..hb*8+23} (col-major weight rows)
__device__ inline v16bf loadB_frag(const unsigned short* row, int kk, int hb) {
    const unsigned short* p = row + kk + hb * 8;
    union { uint4 u[2]; v16bf v; } t;
    t.u[0] = *reinterpret_cast<const uint4*>(p);
    t.u[1] = *reinterpret_cast<const uint4*>(p + 16);
    return t.v;
}

#define WMMA_BF16(A, B, C) \
    __builtin_amdgcn_wmma_f32_16x16x32_bf16(false, (A), false, (B), (short)0, (C), false, false)

// ---- one-time kernels ----------------------------------------------------

// bW[col*KTOT + seg*1024 + k] = bf16(W_seg[k*4096 + col]); seg order: Wh, Wattn, Wx
__global__ __launch_bounds__(256) void convert_weights(
    const float* __restrict__ Wh, const float* __restrict__ Wattn,
    const float* __restrict__ Wx, unsigned short* __restrict__ bW)
{
    size_t idx = (size_t)blockIdx.x * 256 + threadIdx.x;
    const size_t SEG = (size_t)1024 * NC;
    if (idx >= 3 * SEG) return;
    int seg = (int)(idx / SEG);
    size_t rem = idx % SEG;
    int k   = (int)(rem / NC);
    int col = (int)(rem % NC);
    const float* src = (seg == 0) ? Wh : (seg == 1) ? Wattn : Wx;
    bW[(size_t)col * KTOT + (size_t)seg * 1024 + k] = f2bf(src[rem]);
}

// h0 = c0 = mean over 16 spatial slots of A
__global__ __launch_bounds__(256) void init_state(
    const float* __restrict__ A, float* __restrict__ h, float* __restrict__ c)
{
    int idx = blockIdx.x * 256 + threadIdx.x;
    if (idx >= NB * HH) return;
    const float* af = A + (size_t)idx * 16;
    float s = 0.f;
    #pragma unroll
    for (int k = 0; k < 16; ++k) s += af[k];
    s *= (1.0f / 16.0f);
    h[idx] = s;
    c[idx] = s;
}

// ---- per-step kernels ----------------------------------------------------

// scores -> softmax -> attn; also packs act = bf16([h | attn | x_t]) row n
__global__ __launch_bounds__(256) void attn_kernel(
    const float* __restrict__ A, const float* __restrict__ x,
    const float* __restrict__ h, unsigned short* __restrict__ act, int t)
{
    __shared__ float red[16 * 256];
    __shared__ float sm[16];
    int n = blockIdx.x;
    int tid = threadIdx.x;
    const float* Af = A + (size_t)n * HH * 16;
    const float* hn = h + (size_t)n * HH;

    float p[16];
    #pragma unroll
    for (int k = 0; k < 16; ++k) p[k] = 0.f;
    for (int hh = tid; hh < HH; hh += 256) {
        float hv = hn[hh];
        const float* af = Af + (size_t)hh * 16;
        #pragma unroll
        for (int k = 0; k < 16; ++k) p[k] += hv * af[k];
    }
    #pragma unroll
    for (int k = 0; k < 16; ++k) red[k * 256 + tid] = p[k];
    __syncthreads();
    if (tid < 16) {
        float s = 0.f;
        for (int i = 0; i < 256; ++i) s += red[tid * 256 + i];
        sm[tid] = s * 0.03125f;          // 1/sqrt(1024)
    }
    __syncthreads();
    if (tid == 0) {
        float mx = sm[0];
        #pragma unroll
        for (int k = 1; k < 16; ++k) mx = fmaxf(mx, sm[k]);
        float e[16]; float se = 0.f;
        #pragma unroll
        for (int k = 0; k < 16; ++k) { e[k] = __expf(sm[k] - mx); se += e[k]; }
        float inv = 1.f / se;
        #pragma unroll
        for (int k = 0; k < 16; ++k) sm[k] = e[k] * inv;
    }
    __syncthreads();

    unsigned short* actRow = act + (size_t)n * KTOT;
    const float* xr = x + ((size_t)n * TT + t) * DD;
    for (int hh = tid; hh < HH; hh += 256) {
        const float* af = Af + (size_t)hh * 16;
        float s = 0.f;
        #pragma unroll
        for (int k = 0; k < 16; ++k) s += af[k] * sm[k];
        actRow[hh]        = f2bf(hn[hh]);   // segment 0: h       (matches Wh rows)
        actRow[1024 + hh] = f2bf(s);        // segment 1: attn    (matches Wattn rows)
        actRow[2048 + hh] = f2bf(xr[hh]);   // segment 2: x_t     (matches Wx rows)
    }
}

// v = act(512x3072 bf16) @ W(3072x4096 bf16), f32 accumulate via WMMA.
// WG = 8 waves, tile 128(M) x 128(N); each wave: 32(M) x 64(N) = 8 accumulators.
// Per K-iter: 12 x b128 loads feed 8 WMMAs (A frags reused 4x, B frags 2x).
__global__ __launch_bounds__(256) void gemm_kernel(
    const unsigned short* __restrict__ act,
    const unsigned short* __restrict__ bW,
    float* __restrict__ v)
{
    int tid  = threadIdx.x;
    int wave = tid >> 5;
    int lane = tid & 31;
    int r    = lane & 15;
    int hb   = lane >> 4;
    int waveM = wave & 3;       // 4 waves along M -> 128 rows
    int waveN = wave >> 2;      // 2 waves along N -> 128 cols
    int m0 = blockIdx.y * 128 + waveM * 32;
    int n0 = blockIdx.x * 128 + waveN * 64;

    const unsigned short* aRow0 = act + (size_t)(m0 +  0 + r) * KTOT;
    const unsigned short* aRow1 = act + (size_t)(m0 + 16 + r) * KTOT;
    const unsigned short* bRow0 = bW  + (size_t)(n0 +  0 + r) * KTOT;
    const unsigned short* bRow1 = bW  + (size_t)(n0 + 16 + r) * KTOT;
    const unsigned short* bRow2 = bW  + (size_t)(n0 + 32 + r) * KTOT;
    const unsigned short* bRow3 = bW  + (size_t)(n0 + 48 + r) * KTOT;

    v8f z = {0.f,0.f,0.f,0.f,0.f,0.f,0.f,0.f};
    v8f acc00 = z, acc01 = z, acc02 = z, acc03 = z;   // rows m0..m0+15
    v8f acc10 = z, acc11 = z, acc12 = z, acc13 = z;   // rows m0+16..m0+31

    for (int kk = 0; kk < KTOT; kk += 32) {
        v16bf a0 = loadA_frag(aRow0, kk, hb);
        v16bf a1 = loadA_frag(aRow1, kk, hb);
        v16bf b0 = loadB_frag(bRow0, kk, hb);
        v16bf b1 = loadB_frag(bRow1, kk, hb);
        v16bf b2 = loadB_frag(bRow2, kk, hb);
        v16bf b3 = loadB_frag(bRow3, kk, hb);
        acc00 = WMMA_BF16(a0, b0, acc00);
        acc10 = WMMA_BF16(a1, b0, acc10);
        acc01 = WMMA_BF16(a0, b1, acc01);
        acc11 = WMMA_BF16(a1, b1, acc11);
        acc02 = WMMA_BF16(a0, b2, acc02);
        acc12 = WMMA_BF16(a1, b2, acc12);
        acc03 = WMMA_BF16(a0, b3, acc03);
        acc13 = WMMA_BF16(a1, b3, acc13);
    }

    // C/D layout: acc[i] in lane l -> row mbase + i + 8*(l>>4), col nbase + (l&15)
    float* vb0 = v + (size_t)(m0 +  0 + hb * 8) * NC + n0 + r;
    float* vb1 = v + (size_t)(m0 + 16 + hb * 8) * NC + n0 + r;
    #pragma unroll
    for (int i = 0; i < 8; ++i) {
        vb0[(size_t)i * NC +  0] = acc00[i];
        vb0[(size_t)i * NC + 16] = acc01[i];
        vb0[(size_t)i * NC + 32] = acc02[i];
        vb0[(size_t)i * NC + 48] = acc03[i];
        vb1[(size_t)i * NC +  0] = acc10[i];
        vb1[(size_t)i * NC + 16] = acc11[i];
        vb1[(size_t)i * NC + 32] = acc12[i];
        vb1[(size_t)i * NC + 48] = acc13[i];
    }
}

// gates + state update + output write for timestep t
__global__ __launch_bounds__(256) void gate_kernel(
    const float* __restrict__ v, const float* __restrict__ b,
    float* __restrict__ h, float* __restrict__ c,
    float* __restrict__ out, int t)
{
    int idx = blockIdx.x * 256 + threadIdx.x;
    if (idx >= NB * HH) return;
    int n = idx >> 10;
    int j = idx & 1023;
    const float* vr = v + (size_t)n * NC;
    float vi = vr[j]        + b[j];
    float vf = vr[1024 + j] + b[1024 + j];
    float vo = vr[2048 + j] + b[2048 + j];
    float vg = vr[3072 + j] + b[3072 + j];
    float ig = 1.f / (1.f + __expf(-vi));
    float fg = 1.f / (1.f + __expf(-vf));
    float og = 1.f / (1.f + __expf(-vo));
    float gg = tanhf(vg);
    float cn = fg * c[idx] + ig * gg;
    c[idx] = cn;
    float hn = og * tanhf(cn);
    h[idx] = hn;
    out[((size_t)n * TT + t) * HH + j] = hn;
}

// ---- host orchestration --------------------------------------------------

extern "C" void kernel_launch(void* const* d_in, const int* in_sizes, int n_in,
                              void* d_out, int out_size, void* d_ws, size_t ws_size,
                              hipStream_t stream) {
    const float* x     = (const float*)d_in[0];   // (N,T,D)
    const float* A     = (const float*)d_in[1];   // (N,H,4,4)
    const float* Wx    = (const float*)d_in[2];   // (D,4H)
    const float* Wh    = (const float*)d_in[3];   // (H,4H)
    const float* Wattn = (const float*)d_in[4];   // (H,4H)
    const float* b     = (const float*)d_in[5];   // (4H)
    float* out = (float*)d_out;                   // (N,T,H)

    char* ws = (char*)d_ws;
    size_t off = 0;
    auto carve = [&](size_t bytes) -> void* {
        void* p = ws + off;
        off = (off + bytes + 255) & ~(size_t)255;
        return p;
    };
    unsigned short* bW  = (unsigned short*)carve((size_t)KTOT * NC * 2);   // 24 MB
    unsigned short* act = (unsigned short*)carve((size_t)NB * KTOT * 2);   //  3 MB
    float* v = (float*)carve((size_t)NB * NC * 4);                         //  8 MB
    float* h = (float*)carve((size_t)NB * HH * 4);                         //  2 MB
    float* c = (float*)carve((size_t)NB * HH * 4);                         //  2 MB

    {
        size_t tot = (size_t)3 * 1024 * NC;
        convert_weights<<<(unsigned)((tot + 255) / 256), 256, 0, stream>>>(Wh, Wattn, Wx, bW);
    }
    init_state<<<(NB * HH + 255) / 256, 256, 0, stream>>>(A, h, c);

    for (int t = 0; t < TT; ++t) {
        attn_kernel<<<NB, 256, 0, stream>>>(A, x, h, act, t);
        gemm_kernel<<<dim3(NC / 128, NB / 128), 256, 0, stream>>>(act, bW, v);
        gate_kernel<<<(NB * HH) / 256, 256, 0, stream>>>(v, b, h, c, out, t);
    }
}
